// GIN_31104153157817
// MI455X (gfx1250) — matrix operand order repaired and checked by reading, not compiled, not run on hardware
//
#include <hip/hip_runtime.h>
#include <hip/hip_bf16.h>

#define N_NODES 50000
#define N_EDGES 800000
#define N_GRAPH 500
#define HDIM    128
#define ODIM    10

typedef __attribute__((ext_vector_type(16))) __bf16 v16bf;
typedef __attribute__((ext_vector_type(8)))  float  v8f;

union Frag16 { unsigned int u[8]; v16bf v; };

#if defined(__HIP_DEVICE_COMPILE__) && __has_builtin(__builtin_amdgcn_tensor_load_to_lds) && \
    __has_builtin(__builtin_amdgcn_s_wait_tensorcnt)
#define USE_TDM 1
typedef unsigned int u32x4 __attribute__((ext_vector_type(4)));
typedef int          i32x8 __attribute__((ext_vector_type(8)));
typedef int          i32x4 __attribute__((ext_vector_type(4)));
#else
#define USE_TDM 0
#endif

__device__ __forceinline__ unsigned short f2bf(float f) {
  unsigned int u = __float_as_uint(f);
  u += 0x7fffu + ((u >> 16) & 1u);          // round-to-nearest-even
  return (unsigned short)(u >> 16);
}

// one-time f32 -> bf16 weight conversion into workspace (weights stay L2-resident)
__global__ void w2bf_kernel(const float* __restrict__ src, unsigned short* __restrict__ dst, int n) {
  int i = blockIdx.x * blockDim.x + threadIdx.x;
  if (i < n) dst[i] = f2bf(src[i]);
}

// out = (1+eps) * h   (initializes aggregation buffer; eps read from device scalar)
__global__ void pre_init_kernel(const float* __restrict__ h, float* __restrict__ out,
                                const float* __restrict__ epsPtr, int total) {
  int i = blockIdx.x * blockDim.x + threadIdx.x;
  if (i < total) out[i] = (1.0f + *epsPtr) * h[i];
}

// agg[dst] += h[src] over edges; one wave32 per edge, float4 per lane (32*4 = 128 feats)
__global__ void scatter_kernel(const float* __restrict__ h, const int* __restrict__ ei,
                               float* __restrict__ agg) {
  int e = blockIdx.x * 8 + (threadIdx.x >> 5);
  if (e >= N_EDGES) return;
  int lane = threadIdx.x & 31;
  int src = ei[e];
  int dst = ei[N_EDGES + e];
  float4 v = ((const float4*)(h + (size_t)src * HDIM))[lane];
  float* d = agg + (size_t)dst * HDIM + lane * 4;
  atomicAdd(d + 0, v.x); atomicAdd(d + 1, v.y);
  atomicAdd(d + 2, v.z); atomicAdd(d + 3, v.w);
}

// C = epilogue(A[64rows x128] @ W[128x128]); bf16 WMMA, f32 accumulate.
// W comes in as pre-converted bf16; staged into LDS by the Tensor Data Mover
// (with hardware LDS padding matching our 68-dword anti-bank-conflict stride).
// doBN==0: relu(C+bias).  doBN==1: BN(relu(C+bias)) with running stats.
__global__ void gemm128_kernel(const float* __restrict__ A,
                               const unsigned short* __restrict__ Wbf,
                               const float* __restrict__ bias,
                               const float* __restrict__ gamma, const float* __restrict__ beta,
                               const float* __restrict__ rmean, const float* __restrict__ rvar,
                               float* __restrict__ out, int doBN) {
  __shared__ __align__(16) unsigned short Al[64 * 130];   // A tile, row stride 65 dwords
  __shared__ __align__(16) unsigned short Wl[128 * 136];  // W, row stride 68 dwords (TDM-padded)
  int tid = threadIdx.x;                                  // 128 threads = 4 waves
  long rowTile = (long)blockIdx.x * 64;

#if USE_TDM
  // wave 0 kicks off the 32 KB W DMA; it overlaps with the A-tile staging below.
  if (tid < 32) {
    unsigned long long ga = (unsigned long long)(const void*)Wbf;
    unsigned int lds_off = (unsigned int)(unsigned long long)(const void*)&Wl[0];
    u32x4 g0;
    g0.x = 1u;                                              // count = 1 (valid D#)
    g0.y = lds_off;                                         // lds_addr
    g0.z = (unsigned int)(ga & 0xffffffffull);              // global_addr[31:0]
    g0.w = (unsigned int)((ga >> 32) & 0x01ffffffull)       // global_addr[56:32]
         | (2u << 30);                                      // type = 2 ("image")
    i32x8 g1;
    g1[0] = 0x07510000;  // data_size=2B | pad_enable | pad_interval=5(64dw) | pad_amount=3(4dw)
    g1[1] = 0x00800000;  // tensor_dim0 = 128
    g1[2] = 0x00800000;  // tensor_dim1 = 128
    g1[3] = 0x00800000;  // tile_dim0   = 128
    g1[4] = 0x00000080;  // tile_dim1   = 128, tile_dim2 = 0
    g1[5] = 0x00000080;  // tensor_dim0_stride = 128
    g1[6] = 0; g1[7] = 0;
    i32x4 z4 = {0, 0, 0, 0};
    i32x8 z8 = {0, 0, 0, 0, 0, 0, 0, 0};
    __builtin_amdgcn_tensor_load_to_lds(g0, g1, z4, z4, z8, 0);
  }
#endif

  // stage A (64x128) as bf16, zero-pad past N (overlaps with TDM W transfer)
  for (int idx = tid; idx < 64 * HDIM; idx += 128) {
    int r = idx >> 7, c = idx & 127;
    long gr = rowTile + r;
    float v = (gr < N_NODES) ? A[gr * HDIM + c] : 0.0f;
    Al[r * 130 + c] = f2bf(v);
  }

#if USE_TDM
  if (tid < 32) __builtin_amdgcn_s_wait_tensorcnt(0);   // wave 0 waits for DMA done
#else
  for (int idx = tid; idx < HDIM * HDIM; idx += 128) {  // fallback: plain bf16 copy
    int r = idx >> 7, c = idx & 127;
    Wl[r * 136 + c] = Wbf[idx];
  }
#endif
  __syncthreads();

  int wave = tid >> 5, lane = tid & 31;
  int laneHi = lane >> 4, m = lane & 15;
  int rowL = wave * 16 + m;                 // local A row this lane supplies
  const unsigned int* Ad = (const unsigned int*)Al;
  const unsigned int* Wd = (const unsigned int*)Wl;

  v8f acc[8] = {};
  #pragma unroll
  for (int kc = 0; kc < 4; ++kc) {          // K = 128 in chunks of 32
    Frag16 a;
    #pragma unroll
    for (int j = 0; j < 8; ++j) {
      // ISA 16-bit A 16x32 layout: lane<16 -> K pairs {2j | 16+2(j-4)}, lane>=16 -> +8
      int kl = (laneHi ? 8 : 0) + ((j & 4) ? 16 : 0) + ((j & 3) << 1);
      a.u[j] = Ad[rowL * 65 + ((kc * 32 + kl) >> 1)];
    }
    #pragma unroll
    for (int ct = 0; ct < 8; ++ct) {        // 8 column tiles of 16
      Frag16 b;
      int ncol = ct * 16 + m;               // B layout: lane&15 = N column, K order as A
      #pragma unroll
      for (int j = 0; j < 8; ++j) {
        int k0 = kc * 32 + (laneHi ? 8 : 0) + ((j & 4) ? 16 : 0) + ((j & 3) << 1);
        b.u[j] = (unsigned int)Wl[k0 * 136 + ncol] | ((unsigned int)Wl[(k0 + 1) * 136 + ncol] << 16);
      }
      acc[ct] = __builtin_amdgcn_wmma_f32_16x16x32_bf16(
          false, a.v, false, b.v, (short)0, acc[ct], false, false);
    }
  }

  // C/D layout: VGPR r -> M = 8*laneHi + r, N = lane&15
  long rowBase = rowTile + wave * 16 + laneHi * 8;
  #pragma unroll
  for (int ct = 0; ct < 8; ++ct) {
    int c = ct * 16 + m;
    float bv = bias[c];
    float sc = 1.0f, sh = 0.0f;
    if (doBN) {
      float s = gamma[c] * rsqrtf(rvar[c] + 1e-5f);
      sc = s; sh = beta[c] - rmean[c] * s;
    }
    #pragma unroll
    for (int r = 0; r < 8; ++r) {
      long row = rowBase + r;
      if (row < N_NODES) {
        float v = fmaxf(acc[ct][r] + bv, 0.0f);
        if (doBN) v = v * sc + sh;
        out[row * HDIM + c] = v;
      }
    }
  }
}

__global__ void pool_zero_kernel(float* __restrict__ hg, float* __restrict__ cnt) {
  int i = blockIdx.x * blockDim.x + threadIdx.x;
  if (i < N_GRAPH * HDIM) hg[i] = 0.0f;
  if (i < N_GRAPH) cnt[i] = 0.0f;
}

// per-graph sum + count; one wave32 per node
__global__ void pool_acc_kernel(const float* __restrict__ h, const int* __restrict__ batch,
                                float* __restrict__ hg, float* __restrict__ cnt) {
  int nid = blockIdx.x * 8 + (threadIdx.x >> 5);
  if (nid >= N_NODES) return;
  int lane = threadIdx.x & 31;
  int g = batch[nid];
  float4 v = ((const float4*)(h + (size_t)nid * HDIM))[lane];
  float* d = hg + (size_t)g * HDIM + lane * 4;
  atomicAdd(d + 0, v.x); atomicAdd(d + 1, v.y);
  atomicAdd(d + 2, v.z); atomicAdd(d + 3, v.w);
  if (lane == 0) atomicAdd(cnt + g, 1.0f);
}

// mean -> relu(lin1) -> lin2 -> log_softmax; one block per graph (tiny: 500 x ~17 KFLOP)
__global__ void head_kernel(const float* __restrict__ hg, const float* __restrict__ cnt,
                            const float* __restrict__ W1, const float* __restrict__ b1,
                            const float* __restrict__ W2, const float* __restrict__ b2,
                            float* __restrict__ out) {
  __shared__ float s[HDIM];
  __shared__ float t[HDIM];
  __shared__ float lg[ODIM];
  int g = blockIdx.x, tid = threadIdx.x;
  float c = fmaxf(cnt[g], 1.0f);
  s[tid] = hg[(size_t)g * HDIM + tid] / c;
  __syncthreads();
  float a = b1[tid];
  #pragma unroll 4
  for (int k = 0; k < HDIM; ++k) a += s[k] * W1[k * HDIM + tid];
  t[tid] = fmaxf(a, 0.0f);
  __syncthreads();
  if (tid < ODIM) {
    float o = b2[tid];
    for (int k = 0; k < HDIM; ++k) o += t[k] * W2[k * ODIM + tid];
    lg[tid] = o;
  }
  __syncthreads();
  if (tid == 0) {
    float mx = lg[0];
    for (int i = 1; i < ODIM; ++i) mx = fmaxf(mx, lg[i]);
    float se = 0.0f;
    for (int i = 0; i < ODIM; ++i) se += __expf(lg[i] - mx);
    float lse = mx + __logf(se);
    for (int i = 0; i < ODIM; ++i) out[g * ODIM + i] = lg[i] - lse;
  }
}

extern "C" void kernel_launch(void* const* d_in, const int* in_sizes, int n_in,
                              void* d_out, int out_size, void* d_ws, size_t ws_size,
                              hipStream_t stream) {
  const float* x      = (const float*)d_in[0];
  const int*   ei     = (const int*)  d_in[1];
  const int*   batch  = (const int*)  d_in[2];
  const float* c1_W1  = (const float*)d_in[3];
  const float* c1_b1  = (const float*)d_in[4];
  const float* c1_W2  = (const float*)d_in[5];
  const float* c1_b2  = (const float*)d_in[6];
  const float* c1_g   = (const float*)d_in[7];
  const float* c1_be  = (const float*)d_in[8];
  const float* c1_rm  = (const float*)d_in[9];
  const float* c1_rv  = (const float*)d_in[10];
  const float* c1_eps = (const float*)d_in[11];
  const float* Ws1    = (const float*)d_in[12];
  const float* bs1    = (const float*)d_in[13];
  const float* Ws2    = (const float*)d_in[14];
  const float* bs2    = (const float*)d_in[15];
  const float* gs     = (const float*)d_in[16];
  const float* bes    = (const float*)d_in[17];
  const float* rms    = (const float*)d_in[18];
  const float* rvs    = (const float*)d_in[19];
  const float* epss   = (const float*)d_in[20];
  const float* lin1_W = (const float*)d_in[21];
  const float* lin1_b = (const float*)d_in[22];
  const float* lin2_W = (const float*)d_in[23];
  const float* lin2_b = (const float*)d_in[24];
  float* out = (float*)d_out;

  const size_t NF = (size_t)N_NODES * HDIM;
  float* B0  = (float*)d_ws;            // activation ping-pong buffers (L2-resident)
  float* B1  = B0 + NF;
  float* B2  = B1 + NF;
  float* hg  = B2 + NF;
  float* cnt = hg + (size_t)N_GRAPH * HDIM;
  // bf16 weight cache, 256B-aligned for the TDM
  size_t wbOff = (((size_t)(char*)(cnt + N_GRAPH) - (size_t)(char*)d_ws) + 255) & ~(size_t)255;
  unsigned short* WB = (unsigned short*)((char*)d_ws + wbOff);
  const int WSZ = HDIM * HDIM;

  dim3 blk256(256), blk128(128);
  int preGrid = (int)((NF + 255) / 256);
  int scGrid  = (N_EDGES + 7) / 8;
  int gmGrid  = (N_NODES + 63) / 64;
  int paGrid  = (N_NODES + 7) / 8;
  int pzGrid  = (N_GRAPH * HDIM + 255) / 256;
  int wcGrid  = (WSZ + 255) / 256;

  // ---- one-time bf16 weight conversion (6 x 16K elements, trivial) ----
  w2bf_kernel<<<wcGrid, blk256, 0, stream>>>(c1_W1,      WB + 0 * WSZ, WSZ);
  w2bf_kernel<<<wcGrid, blk256, 0, stream>>>(c1_W2,      WB + 1 * WSZ, WSZ);
  w2bf_kernel<<<wcGrid, blk256, 0, stream>>>(Ws1,        WB + 2 * WSZ, WSZ);
  w2bf_kernel<<<wcGrid, blk256, 0, stream>>>(Ws2,        WB + 3 * WSZ, WSZ);
  w2bf_kernel<<<wcGrid, blk256, 0, stream>>>(Ws1 + WSZ,  WB + 4 * WSZ, WSZ);
  w2bf_kernel<<<wcGrid, blk256, 0, stream>>>(Ws2 + WSZ,  WB + 5 * WSZ, WSZ);

  // ---- layer 0 (h = x) : pre=B0, mid=B1, h1=B0 ----
  pre_init_kernel<<<preGrid, blk256, 0, stream>>>(x, B0, c1_eps, (int)NF);
  scatter_kernel <<<scGrid, blk256, 0, stream>>>(x, ei, B0);
  gemm128_kernel <<<gmGrid, blk128, 0, stream>>>(B0, WB + 0 * WSZ, c1_b1, c1_g, c1_be, c1_rm, c1_rv, B1, 0);
  gemm128_kernel <<<gmGrid, blk128, 0, stream>>>(B1, WB + 1 * WSZ, c1_b2, c1_g, c1_be, c1_rm, c1_rv, B0, 1);

  // ---- layer 1 : h=B0, pre=B1, mid=B2, h2=B1 ----
  pre_init_kernel<<<preGrid, blk256, 0, stream>>>(B0, B1, epss + 0, (int)NF);
  scatter_kernel <<<scGrid, blk256, 0, stream>>>(B0, ei, B1);
  gemm128_kernel <<<gmGrid, blk128, 0, stream>>>(B1, WB + 2 * WSZ, bs1, gs, bes, rms, rvs, B2, 0);
  gemm128_kernel <<<gmGrid, blk128, 0, stream>>>(B2, WB + 3 * WSZ, bs2, gs, bes, rms, rvs, B1, 1);

  // ---- layer 2 : h=B1, pre=B2, mid=B0, h3=B2 ----
  pre_init_kernel<<<preGrid, blk256, 0, stream>>>(B1, B2, epss + 1, (int)NF);
  scatter_kernel <<<scGrid, blk256, 0, stream>>>(B1, ei, B2);
  gemm128_kernel <<<gmGrid, blk128, 0, stream>>>(B2, WB + 4 * WSZ, bs1 + HDIM,
                                                 gs + HDIM, bes + HDIM, rms + HDIM, rvs + HDIM, B0, 0);
  gemm128_kernel <<<gmGrid, blk128, 0, stream>>>(B0, WB + 5 * WSZ, bs2 + HDIM,
                                                 gs + HDIM, bes + HDIM, rms + HDIM, rvs + HDIM, B2, 1);

  // ---- readout ----
  pool_zero_kernel<<<pzGrid, blk256, 0, stream>>>(hg, cnt);
  pool_acc_kernel <<<paGrid, blk256, 0, stream>>>(B2, batch, hg, cnt);
  head_kernel     <<<N_GRAPH, blk128, 0, stream>>>(hg, cnt, lin1_W, lin1_b, lin2_W, lin2_b, out);

  (void)in_sizes; (void)n_in; (void)out_size; (void)ws_size;
}
